// GCN_15753940042077
// MI455X (gfx1250) — matrix operand order repaired and checked by reading, not compiled, not run on hardware
//
#include <hip/hip_runtime.h>
#include <hip/hip_bf16.h>

// ---------------------------------------------------------------------------
// GCN forward for MI455X (gfx1250, wave32, WMMA).
//   prep  : W1, W2 pre-swizzled to bf16 WMMA-fragment order (per launch)
//   layer1: support1 = x @ W1                       (bf16 WMMA, f32 accum)
//           agg1     = scatter_add(ew*support1[src]) -> dst  (f32 L2 atomics)
//   layer2: A = dropout(relu(agg1+b1)) fused into GEMM staging
//           support2 = A @ W2                       (bf16 WMMA, N=40 masked)
//           agg2     = scatter_add(ew*support2[src]) -> dst
//   out    = log_softmax(agg2 + b2)                 (one wave32 per row)
// ---------------------------------------------------------------------------

typedef __attribute__((ext_vector_type(16))) __bf16 bf16x16;
typedef __attribute__((ext_vector_type(8)))  float  f32x8;
typedef __attribute__((ext_vector_type(4)))  float  f32x4;

#define D_IN  512
#define D_HID 256
#define D_OUT 40
#define D_OUTP 48            // D_OUT padded to 3 WMMA N-tiles

__device__ __forceinline__ unsigned pack_bf16x2(float lo, float hi) {
    union { __bf16 b[2]; unsigned u; } p;
    p.b[0] = (__bf16)lo;
    p.b[1] = (__bf16)hi;
    return p.u;
}

// A fragment (16x32 bf16) from an LDS tile stored row-major [16][32] bf16.
// ISA 7.12.2: lane m=l&15, half=l>>4; elems 0..7 -> K=8*half+0..7,
// elems 8..15 -> K=16+8*half+0..7  => two contiguous 16B chunks per lane.
__device__ __forceinline__ bf16x16 load_a_frag(const unsigned short* As, int lane) {
    const int m    = lane & 15;
    const int half = lane >> 4;
    const char* base = (const char*)As + m * 64;           // 64B per row
    union { bf16x16 v; uint4 q[2]; } u;
    u.q[0] = *(const uint4*)(base + half * 16);            // ds_load_b128
    u.q[1] = *(const uint4*)(base + 32 + half * 16);       // ds_load_b128
    return u.v;
}

// B fragment from pre-swizzled weights: layout [kb][t][lane(32)][e(16)] bf16,
// i.e. 512 bf16 (1KB) per (kb,t) block, 32B contiguous per lane.
__device__ __forceinline__ bf16x16 load_b_frag(const unsigned short* Wf,
                                               int blockIdxKbT, int lane) {
    const uint4* p = (const uint4*)(Wf + ((size_t)blockIdxKbT * 32 + lane) * 16);
    union { bf16x16 v; uint4 q[2]; } u;
    u.q[0] = p[0];                                         // global_load_b128
    u.q[1] = p[1];                                         // global_load_b128
    return u.v;
}

// ---------------------------------------------------------------------------
// Weight swizzle: W[K,N] fp32 (row-major) -> Wf bf16 fragment order
// [kb = K/32][t = Npad/16][lane][e], with B-fragment mapping
//   lane = (n%16) + 16*half, e<8: K = kb*32 + 8*half + e, e>=8: K = kb*32 + 16 + 8*half + (e-8), n = t*16 + n%16.
// One thread per (kb,t,lane,epair); cols >= N are zero-padded.
// ---------------------------------------------------------------------------
__global__ __launch_bounds__(256) void gcn_swizzle_w(const float* __restrict__ W,
                                                     unsigned short* __restrict__ Wf,
                                                     int K, int N, int Npad) {
    const int NT = Npad / 16;
    const int total = (K / 32) * NT * 32 * 8;
    const int tid = blockIdx.x * 256 + threadIdx.x;
    if (tid >= total) return;
    const int epair = tid & 7;
    const int lane  = (tid >> 3) & 31;
    const int t     = (tid >> 8) % NT;
    const int kb    = tid / (256 * NT);
    const int half  = lane >> 4;
    const int n     = t * 16 + (lane & 15);
    const int k0    = kb * 32 + 8 * half + 2 * (epair & 3) + ((epair >> 2) << 4);
    const float a = (n < N) ? W[(size_t)k0 * N + n] : 0.0f;
    const float b = (n < N) ? W[(size_t)(k0 + 1) * N + n] : 0.0f;
    ((unsigned*)Wf)[tid] = pack_bf16x2(a, b);
}

// ---------------------------------------------------------------------------
// GEMM1: Y[M,256] = X[M,512] @ W1f.
// grid = M/16, block = 128 (4 waves). Block covers all 256 columns:
// wave w owns cols [64w,64w+64) as 4 N-tiles sharing one A fragment.
// Inner loop: 2 ds_load_b128 (A) + 8 global_load_b128 (B) + 4 WMMA.
// ---------------------------------------------------------------------------
__global__ __launch_bounds__(128) void gcn_gemm1(const float* __restrict__ X,
                                                 const unsigned short* __restrict__ Wf,
                                                 float* __restrict__ Y,
                                                 int M, int K, int N) {
    __shared__ __align__(16) unsigned short As[16 * 32];

    const int lane  = threadIdx.x & 31;
    const int wave  = threadIdx.x >> 5;
    const int m0    = blockIdx.x * 16;
    const int half  = lane >> 4;
    const int ncol0 = wave * 64 + (lane & 15);
    const int NT    = N / 16;

    const int srow  = threadIdx.x >> 3;          // 0..15
    const int skcol = (threadIdx.x & 7) * 4;     // 0,4,..,28
    const float* __restrict__ xstage =
        X + (size_t)min(m0 + srow, M - 1) * K + skcol;

    f32x8 acc[4] = {};
    for (int kk = 0; kk < K; kk += 32) {
        __syncthreads();
        // x streamed once: non-temporal so it doesn't evict L2 working set
        const f32x4 v = __builtin_nontemporal_load((const f32x4*)(xstage + kk));
        *(uint2*)((char*)As + srow * 64 + skcol * 2) =
            make_uint2(pack_bf16x2(v[0], v[1]), pack_bf16x2(v[2], v[3]));
        __syncthreads();

        const bf16x16 a = load_a_frag(As, lane);
        const int kb = kk >> 5;
#pragma unroll
        for (int t = 0; t < 4; ++t) {
            const bf16x16 b = load_b_frag(Wf, kb * NT + wave * 4 + t, lane);
            acc[t] = __builtin_amdgcn_wmma_f32_16x16x32_bf16(
                false, a, false, b, (short)0, acc[t], false, false);
        }
    }
#pragma unroll
    for (int t = 0; t < 4; ++t) {
        float* __restrict__ yout =
            Y + (size_t)(m0 + 8 * half) * N + ncol0 + t * 16;
#pragma unroll
        for (int r = 0; r < 8; ++r) yout[(size_t)r * N] = acc[t][r];
    }
}

// ---------------------------------------------------------------------------
// GEMM2 (fused): A = dropout(relu(agg1 + b1)) computed during staging, then
// Y[M,40] = A[M,256] @ W2f (48 padded cols, 3 tiles).
// grid = M/16, block = 128 (4 waves): all waves stage, waves 0..2 compute one
// N-tile each (wave-uniform branch keeps EXEC all-1s for WMMA; barriers
// outside the branch). Stores masked to ncol < 40.
// ---------------------------------------------------------------------------
__global__ __launch_bounds__(128) void gcn_gemm2(const float* __restrict__ AGG,
                                                 const float* __restrict__ b1,
                                                 const unsigned short* __restrict__ Wf,
                                                 float* __restrict__ Y,
                                                 int M, int K, int Nout) {
    __shared__ __align__(16) unsigned short As[16 * 32];

    const int lane  = threadIdx.x & 31;
    const int wave  = threadIdx.x >> 5;
    const int m0    = blockIdx.x * 16;
    const int half  = lane >> 4;
    const int ncol  = wave * 16 + (lane & 15);
    const int NT    = D_OUTP / 16;               // 3

    const int srow  = threadIdx.x >> 3;
    const int skcol = (threadIdx.x & 7) * 4;
    const int arow  = min(m0 + srow, M - 1);
    const float* __restrict__ hstage = AGG + (size_t)arow * K + skcol;

    f32x8 acc = {};
    for (int kk = 0; kk < K; kk += 32) {
        __syncthreads();
        f32x4 v = __builtin_nontemporal_load((const f32x4*)(hstage + kk));
        // fused bias + relu + dropout (deterministic hash, p_keep = 0.5)
#pragma unroll
        for (int j = 0; j < 4; ++j) {
            const int col = kk + skcol + j;
            const float h = fmaxf(v[j] + b1[col], 0.0f);
            unsigned s = (unsigned)((size_t)arow * K + col) * 2654435761u + 42u;
            s ^= s >> 16; s *= 2246822519u;
            s ^= s >> 13; s *= 3266489917u;
            s ^= s >> 16;
            v[j] = (s & 0x80000000u) ? 0.0f : h * 2.0f;
        }
        *(uint2*)((char*)As + srow * 64 + skcol * 2) =
            make_uint2(pack_bf16x2(v[0], v[1]), pack_bf16x2(v[2], v[3]));
        __syncthreads();

        if (wave < 3) {                          // wave-uniform
            const bf16x16 a = load_a_frag(As, lane);
            const bf16x16 b = load_b_frag(Wf, (kk >> 5) * NT + wave, lane);
            acc = __builtin_amdgcn_wmma_f32_16x16x32_bf16(
                false, a, false, b, (short)0, acc, false, false);
        }
    }
    if (wave < 3 && ncol < Nout) {
        float* __restrict__ yout = Y + (size_t)(m0 + 8 * half) * Nout + ncol;
#pragma unroll
        for (int r = 0; r < 8; ++r) yout[(size_t)r * Nout] = acc[r];
    }
}

// ---------------------------------------------------------------------------
// SpMM scatter: agg[dst[e], :] += ew[e] * S[src[e], :]   (float4 per thread)
// Consecutive threads -> consecutive float4 of same edge (coalesced).
// S + agg (~200MB) ~fit the 192MB L2: gathers and the fp32 atomics
// (global_atomic_add_f32) resolve at L2 rate.
// ---------------------------------------------------------------------------
__global__ __launch_bounds__(256) void gcn_spmm(const int* __restrict__ src,
                                                const int* __restrict__ dst,
                                                const float* __restrict__ ew,
                                                const float* __restrict__ S,
                                                float* __restrict__ agg,
                                                long long E, int F4, int F) {
    const long long tid = (long long)blockIdx.x * blockDim.x + threadIdx.x;
    if (tid >= E * (long long)F4) return;
    const long long e  = tid / F4;
    const int       f4 = (int)(tid % F4);
    const float w = ew[e];
    const f32x4 v = *(const f32x4*)(S + (size_t)src[e] * F + f4 * 4);
    float* o = agg + (size_t)dst[e] * F + f4 * 4;
    unsafeAtomicAdd(o + 0, w * v[0]);
    unsafeAtomicAdd(o + 1, w * v[1]);
    unsafeAtomicAdd(o + 2, w * v[2]);
    unsafeAtomicAdd(o + 3, w * v[3]);
}

// ---------------------------------------------------------------------------
// out = log_softmax(agg2 + b2, axis=1). One wave32 per row of 40:
// lane covers col=lane; lanes 0-7 also cover col=lane+32.
// ---------------------------------------------------------------------------
__global__ __launch_bounds__(256) void gcn_logsoftmax(const float* __restrict__ agg,
                                                      const float* __restrict__ b2,
                                                      float* __restrict__ out,
                                                      int M, int F) {
    const int row  = blockIdx.x * 8 + (threadIdx.x >> 5);
    const int lane = threadIdx.x & 31;
    if (row >= M) return;
    const float* a = agg + (size_t)row * F;
    const float x0 = a[lane] + b2[lane];
    const float x1 = (lane < 8) ? (a[lane + 32] + b2[lane + 32]) : -INFINITY;

    float mx = fmaxf(x0, x1);
#pragma unroll
    for (int off = 16; off > 0; off >>= 1) mx = fmaxf(mx, __shfl_xor(mx, off, 32));
    float sum = __expf(x0 - mx) + ((lane < 8) ? __expf(x1 - mx) : 0.0f);
#pragma unroll
    for (int off = 16; off > 0; off >>= 1) sum += __shfl_xor(sum, off, 32);
    const float lse = mx + __logf(sum);

    __builtin_nontemporal_store(x0 - lse, &out[(size_t)row * F + lane]);
    if (lane < 8)
        __builtin_nontemporal_store(x1 - lse, &out[(size_t)row * F + lane + 32]);
}

// ---------------------------------------------------------------------------
extern "C" void kernel_launch(void* const* d_in, const int* in_sizes, int n_in,
                              void* d_out, int out_size, void* d_ws, size_t ws_size,
                              hipStream_t stream) {
    const float* x   = (const float*)d_in[0];
    const int*   src = (const int*)  d_in[1];
    const int*   dst = (const int*)  d_in[2];
    const float* ew  = (const float*)d_in[3];
    const float* W1  = (const float*)d_in[4];
    const float* b1  = (const float*)d_in[5];
    const float* W2  = (const float*)d_in[6];
    const float* b2  = (const float*)d_in[7];
    float* out = (float*)d_out;

    const long long NN = in_sizes[0] / D_IN;   // 100000
    const long long E  = in_sizes[1];          // 1600000

    // Workspace carve-up: ~237MB of f32 buffers + swizzled weights
    float* support1 = (float*)d_ws;                       // NN*256
    float* agg1     = support1 + (size_t)NN * D_HID;      // NN*256
    float* support2 = agg1     + (size_t)NN * D_HID;      // NN*40
    float* agg2     = support2 + (size_t)NN * D_OUT;      // NN*40
    unsigned short* W1f = (unsigned short*)(agg2 + (size_t)NN * D_OUT);
    unsigned short* W2f = W1f + (size_t)D_IN * D_HID;     // 128K bf16 after W1f

    hipMemsetAsync(agg1, 0, (size_t)NN * D_HID * sizeof(float), stream);
    hipMemsetAsync(agg2, 0, (size_t)NN * D_OUT * sizeof(float), stream);

    // Pre-swizzle weights into bf16 WMMA fragment order
    {
        const int t1 = (D_IN / 32) * (D_HID / 16) * 256;   // 65536
        gcn_swizzle_w<<<t1 / 256, 256, 0, stream>>>(W1, W1f, D_IN, D_HID, D_HID);
        const int t2 = (D_HID / 32) * (D_OUTP / 16) * 256; // 6144
        gcn_swizzle_w<<<t2 / 256, 256, 0, stream>>>(W2, W2f, D_HID, D_OUT, D_OUTP);
    }

    // Layer 1 GEMM: (NN x 512) @ (512 x 256)
    gcn_gemm1<<<(unsigned)(NN / 16), 128, 0, stream>>>(x, W1f, support1,
                                                       (int)NN, D_IN, D_HID);

    // Layer 1 SpMM scatter
    {
        const long long work   = E * (D_HID / 4);
        const long long blocks = (work + 255) / 256;
        gcn_spmm<<<(unsigned)blocks, 256, 0, stream>>>(src, dst, ew, support1,
                                                       agg1, E, D_HID / 4, D_HID);
    }

    // Layer 2 GEMM (bias+relu+dropout fused into A staging)
    gcn_gemm2<<<(unsigned)(NN / 16), 128, 0, stream>>>(agg1, b1, W2f, support2,
                                                       (int)NN, D_HID, D_OUT);

    // Layer 2 SpMM scatter
    {
        const long long work   = E * (D_OUT / 4);
        const long long blocks = (work + 255) / 256;
        gcn_spmm<<<(unsigned)blocks, 256, 0, stream>>>(src, dst, ew, support2,
                                                       agg2, E, D_OUT / 4, D_OUT);
    }

    // Bias + log_softmax -> out
    gcn_logsoftmax<<<(unsigned)((NN + 7) / 8), 256, 0, stream>>>(agg2, b2, out,
                                                                 (int)NN, D_OUT);
}
